// CapsuleCo_61160334295614
// MI455X (gfx1250) — matrix operand order repaired and checked by reading, not compiled, not run on hardware
//
#include <hip/hip_runtime.h>
#include <hip/hip_bf16.h>
#include <math.h>

// Capsule routing: B=64, N=1024, I=64, O=64, L=32, ROUTINGS=3
#define BATCH   64
#define IN_N    1024
#define IN_L    64
#define OUT_N   64
#define OUT_L   32
#define OL      (OUT_N * OUT_L)   // 2048
#define EPSV    1e-5f
#define LDSPAD  36                // 64-row tile pad: hi-half lands on disjoint banks

typedef __attribute__((ext_vector_type(16))) __bf16 v16bf;
typedef __attribute__((ext_vector_type(8)))  float  v8f;

__device__ __forceinline__ float4 add4(float4 a, float4 b) {
    float4 r; r.x = a.x + b.x; r.y = a.y + b.y; r.z = a.z + b.z; r.w = a.w + b.w;
    return r;
}

// ---------------------------------------------------------------------------
// Kernel 1: u_hat[b,n,o,l] = sum_i u[b,n,i] * W[n,o,l,i] + bias[n,o,l]
// Per n: GEMM M=64(batch) x N=2048(O*L) x K=64(I), bf16 WMMA, f32 accum.
// Block: 256 threads = 8 waves; wave w -> tile (M-tile = w&3, OL-tile = w>>2)
// Epilogue transposes the 64x32 block tile through LDS so every global write
// is a b128 of a contiguous run (4 threads cover a full 128B row).
// Grid: (OL/32, IN_N)
// ---------------------------------------------------------------------------
__global__ __launch_bounds__(256) void caps_gemm_uhat(
    const float* __restrict__ u, const float* __restrict__ w,
    const float* __restrict__ bias, float* __restrict__ uhat)
{
    const int n    = blockIdx.y;
    const int olc  = blockIdx.x * 32;           // 32-wide OL chunk
    const int tid  = threadIdx.x;
    const int wv   = tid >> 5;
    const int lane = tid & 31;
    const int mt   = wv & 3;                    // batch tile 0..3
    const int nt   = wv >> 2;                   // OL tile 0..1
    const int b0   = mt * 16;
    const int ol0  = olc + nt * 16;
    const int lr   = lane & 15;
    const int hi   = lane >> 4;                 // half-wave select

    // A source: u[b0+lr, n, :]  (row of I=64 floats)
    const float* uRow = u + ((size_t)(b0 + lr) * IN_N + n) * IN_L;
    // B source: w[(n*OL + ol0+lr)*I + i]
    const float* wRow = w + ((size_t)n * OL + ol0 + lr) * IN_L;

    v8f acc = {};
#pragma unroll
    for (int k0 = 0; k0 < IN_L; k0 += 32) {
        v16bf av, bv;
        // A 16-bit 16x32 layout: lane(lr,hi) elems 0..7 -> K = k0+hi*8+j,
        //                        elems 8..15 -> K = k0+16+hi*8+j
        const int abase1 = k0 + hi * 8;
        const int abase2 = abase1 + 16;
        // B layout: same K order as A (elems 0..7 -> k0+hi*8+j, 8..15 -> k0+16+hi*8+j)
        const int bbase = k0 + hi * 8;
#pragma unroll
        for (int j = 0; j < 8; ++j) {
            av[j]     = (__bf16)uRow[abase1 + j];
            av[8 + j] = (__bf16)uRow[abase2 + j];
        }
#pragma unroll
        for (int j = 0; j < 16; ++j) {
            bv[j] = (__bf16)wRow[bbase + j + (j >> 3) * 8];
        }
        acc = __builtin_amdgcn_wmma_f32_16x16x32_bf16(
                  false, av, false, bv, (short)0, acc, false, false);
    }

    // Stage tile (+bias) in LDS: row = batch 0..63, col = ol_local 0..31.
    __shared__ float tile[64 * LDSPAD];
    const float bval = bias[(size_t)n * OL + ol0 + lr];
#pragma unroll
    for (int r = 0; r < 8; ++r) {
        tile[(b0 + hi * 8 + r) * LDSPAD + nt * 16 + lr] = acc[r] + bval;
    }
    __syncthreads();

    // Re-read transposed: thread -> (batch row, 8-wide ol segment); b128 out.
    const int brow = tid >> 2;                  // 0..63
    const int seg  = tid & 3;                   // 0..3 (8 floats each)
    const float* src = &tile[brow * LDSPAD + seg * 8];
    const float4 x0 = *(const float4*)(src);
    const float4 x1 = *(const float4*)(src + 4);
    float* dst = uhat + ((size_t)brow * IN_N + n) * OL + olc + seg * 8;
    *(float4*)(dst)     = x0;
    *(float4*)(dst + 4) = x1;
}

// ---------------------------------------------------------------------------
// Kernel 2: vout[b,o,l] = squash_l( sum_n uhat[b,n,o,l] * (c ? c[b,n,o] : scale) )
// Block: 256 threads = (32 n-groups) x (8 l-quads); all loads/stores b128.
// Grid: BATCH*OUT_N.
// ---------------------------------------------------------------------------
__global__ __launch_bounds__(256) void caps_reduce_squash(
    const float* __restrict__ uhat, const float* __restrict__ c,
    float* __restrict__ vout, float scale)
{
    const int bo = blockIdx.x;          // b*OUT_N + o
    const int b  = bo >> 6;
    const int o  = bo & (OUT_N - 1);
    const int t  = threadIdx.x;
    const int lq = t & 7;               // l = lq*4 .. lq*4+3
    const int ns = t >> 3;              // n-group 0..31

    float4 acc = {0.f, 0.f, 0.f, 0.f};
    if (c) {
        for (int n = ns; n < IN_N; n += 32) {
            const size_t base = (size_t)(b * IN_N + n) * OUT_N + o;
            const float4 x = *(const float4*)(uhat + base * OUT_L + lq * 4);
            const float wgt = c[base];
            acc.x += x.x * wgt; acc.y += x.y * wgt;
            acc.z += x.z * wgt; acc.w += x.w * wgt;
        }
    } else {
        for (int n = ns; n < IN_N; n += 32) {
            const size_t base = (size_t)(b * IN_N + n) * OUT_N + o;
            const float4 x = *(const float4*)(uhat + base * OUT_L + lq * 4);
            acc.x += x.x; acc.y += x.y; acc.z += x.z; acc.w += x.w;
        }
    }

    __shared__ float4 red[256];
    red[t] = acc;
    __syncthreads();
#pragma unroll
    for (int stride = 128; stride >= 8; stride >>= 1) {
        if (t < stride) red[t] = add4(red[t], red[t + stride]);
        __syncthreads();
    }

    if (t < 8) {                        // lanes 0..7 of wave 0 hold l = t*4..t*4+3
        float4 s = red[t];
        if (!c) { s.x *= scale; s.y *= scale; s.z *= scale; s.w *= scale; }
        float ss = s.x * s.x + s.y * s.y + s.z * s.z + s.w * s.w;
        ss += __shfl_xor(ss, 4, 32);
        ss += __shfl_xor(ss, 2, 32);
        ss += __shfl_xor(ss, 1, 32);
        const float nrm = sqrtf(ss);
        const float k = (ss / (1.f + ss)) / (nrm + EPSV);
        float4 vres;
        vres.x = s.x * k; vres.y = s.y * k; vres.z = s.z * k; vres.w = s.w * k;
        *(float4*)(vout + (size_t)bo * OUT_L + t * 4) = vres;
    }
}

// ---------------------------------------------------------------------------
// Kernel 3: per (b,n): a[o] = sum_l uhat[b,n,o,l]*v[b,o,l];
//           bacc[b,n,o] = (first ? 0 : bacc) + a[o];
//           cout[b,n,o] = softmax_o(bacc)
// Block: 64 threads (one per o), grid: BATCH*IN_N.
// ---------------------------------------------------------------------------
__global__ __launch_bounds__(64) void caps_routing_update(
    const float* __restrict__ uhat, const float* __restrict__ v,
    float* __restrict__ bacc, float* __restrict__ cout, int first)
{
    const int bn = blockIdx.x;
    const int b  = bn >> 10;
    const int n  = bn & (IN_N - 1);
    const int o  = threadIdx.x;

    const float4* up = (const float4*)(uhat + ((size_t)(b * IN_N + n) * OUT_N + o) * OUT_L);
    const float4* vp = (const float4*)(v    + ((size_t)(b * OUT_N) + o) * OUT_L);

    float a = 0.f;
#pragma unroll
    for (int j = 0; j < OUT_L / 4; ++j) {
        const float4 uu = up[j], vv = vp[j];
        a += uu.x * vv.x + uu.y * vv.y + uu.z * vv.z + uu.w * vv.w;
    }

    const size_t bi = (size_t)(b * IN_N + n) * OUT_N + o;
    const float bo = first ? a : (bacc[bi] + a);
    bacc[bi] = bo;

    __shared__ float sh[OUT_N];
    sh[o] = bo;
    __syncthreads();
    float mx = -3.4e38f;
#pragma unroll 8
    for (int k = 0; k < OUT_N; ++k) mx = fmaxf(mx, sh[k]);
    const float e = __expf(bo - mx);
    __syncthreads();
    sh[o] = e;
    __syncthreads();
    float sum = 0.f;
#pragma unroll 8
    for (int k = 0; k < OUT_N; ++k) sum += sh[k];
    cout[bi] = e / sum;
}

// ---------------------------------------------------------------------------
// Launch: GEMM -> reduce(uniform) -> route(first) -> reduce(c) -> route ->
//         reduce(c) into d_out.   Matches ROUTINGS=3 exactly.
// ---------------------------------------------------------------------------
extern "C" void kernel_launch(void* const* d_in, const int* in_sizes, int n_in,
                              void* d_out, int out_size, void* d_ws, size_t ws_size,
                              hipStream_t stream) {
    const float* u    = (const float*)d_in[0];   // [B, N, I]
    const float* w    = (const float*)d_in[1];   // [N, O, L, I]
    const float* bias = (const float*)d_in[2];   // [N, O, L]
    float* out = (float*)d_out;                  // [B, O, L]

    float* ws   = (float*)d_ws;
    float* uhat = ws;                                            // 134,217,728 f
    float* v    = uhat + (size_t)BATCH * IN_N * OL;              //     131,072 f
    float* bacc = v    + (size_t)BATCH * OUT_N * OUT_L;          //   4,194,304 f
    float* c    = bacc + (size_t)BATCH * IN_N * OUT_N;           //   4,194,304 f

    // 1) u_hat via bf16 WMMA (f32 accumulate)
    caps_gemm_uhat<<<dim3(OL / 32, IN_N), 256, 0, stream>>>(u, w, bias, uhat);
    // 2) v1 = squash((1/O) * sum_n u_hat)
    caps_reduce_squash<<<BATCH * OUT_N, 256, 0, stream>>>(uhat, nullptr, v, 1.0f / OUT_N);
    // 3) b = <u_hat, v1>; c = softmax_O(b)
    caps_routing_update<<<BATCH * IN_N, 64, 0, stream>>>(uhat, v, bacc, c, 1);
    // 4) v2 = squash(sum_n u_hat * c)
    caps_reduce_squash<<<BATCH * OUT_N, 256, 0, stream>>>(uhat, c, v, 1.0f);
    // 5) b += <u_hat, v2>; c = softmax_O(b)
    caps_routing_update<<<BATCH * IN_N, 64, 0, stream>>>(uhat, v, bacc, c, 0);
    // 6) out = squash(sum_n u_hat * c)
    caps_reduce_squash<<<BATCH * OUT_N, 256, 0, stream>>>(uhat, c, out, 1.0f);
}
